// ParallelExperts_46291157516502
// MI455X (gfx1250) — matrix operand order repaired and checked by reading, not compiled, not run on hardware
//
#include <hip/hip_runtime.h>
#include <hip/hip_bf16.h>
#include <math.h>

// ---------------------------------------------------------------------------
// MoE (ParallelExperts): T=4096 tokens, D=H=2048, E=8, top-4.
// MI455X (gfx1250) strategy: compute-bound (~1.1 TFLOP) vs ~320MB HBM traffic
// -> run both expert GEMMs on the bf16 WMMA matrix cores with fp32 accum.
// 128x128x32 tiles, 8 waves x (32x64) register blocking = 8 WMMA/wave/K-step,
// double-buffered LDS (one barrier per K-step), packed-transposed B staging.
// ---------------------------------------------------------------------------

typedef __bf16 bf16_t;
typedef __attribute__((ext_vector_type(16))) __bf16 v16bf;
typedef __attribute__((ext_vector_type(8)))  __bf16 v8bf;
typedef __attribute__((ext_vector_type(4)))  __bf16 v4bf;
typedef __attribute__((ext_vector_type(8)))  float  v8f;

#define T_TOK 4096
#define DDIM  2048
#define HDIM  2048
#define NEXP  8
#define KTOP  4

// ---------------------------------------------------------------------------
// fp32 -> bf16 bulk convert
// ---------------------------------------------------------------------------
__global__ __launch_bounds__(256) void cvt_f32_to_bf16(
    const float* __restrict__ in, bf16_t* __restrict__ out, int n4)
{
    int i = blockIdx.x * blockDim.x + threadIdx.x;
    int stride = gridDim.x * blockDim.x;
    for (; i < n4; i += stride) {
        float4 v = ((const float4*)in)[i];
        v4bf o = { (bf16_t)v.x, (bf16_t)v.y, (bf16_t)v.z, (bf16_t)v.w };
        *(v4bf*)(out + (size_t)i * 4) = o;
    }
}

// ---------------------------------------------------------------------------
// zero the `final` region of d_out
// ---------------------------------------------------------------------------
__global__ __launch_bounds__(256) void zero_f32(float* __restrict__ p, int n4)
{
    int i = blockIdx.x * blockDim.x + threadIdx.x;
    int stride = gridDim.x * blockDim.x;
    float4 z = make_float4(0.f, 0.f, 0.f, 0.f);
    for (; i < n4; i += stride) ((float4*)p)[i] = z;
}

// ---------------------------------------------------------------------------
// Routing: one wave32 per token. logits = x@Wg + bg; softmax; top-4; renorm.
// ---------------------------------------------------------------------------
__global__ __launch_bounds__(256) void moe_route(
    const float* __restrict__ x, const float* __restrict__ Wg,
    const float* __restrict__ bg, float* __restrict__ logits_out,
    float* __restrict__ comb)
{
    int wid  = threadIdx.x >> 5;
    int lane = threadIdx.x & 31;
    int t = blockIdx.x * 8 + wid;
    if (t >= T_TOK) return;

    const float* xr = x + (size_t)t * DDIM;
    float acc[NEXP];
#pragma unroll
    for (int e = 0; e < NEXP; ++e) acc[e] = 0.f;

    for (int d = lane; d < DDIM; d += 32) {
        float xv = xr[d];
        const float4* wg4 = (const float4*)(Wg + (size_t)d * NEXP);
        float4 w0 = wg4[0], w1 = wg4[1];
        acc[0] += xv * w0.x; acc[1] += xv * w0.y;
        acc[2] += xv * w0.z; acc[3] += xv * w0.w;
        acc[4] += xv * w1.x; acc[5] += xv * w1.y;
        acc[6] += xv * w1.z; acc[7] += xv * w1.w;
    }
#pragma unroll
    for (int e = 0; e < NEXP; ++e) {
#pragma unroll
        for (int off = 16; off > 0; off >>= 1)
            acc[e] += __shfl_xor(acc[e], off, 32);
    }

    if (lane == 0) {
        float lg[NEXP];
        float mx = -3.4e38f;
#pragma unroll
        for (int e = 0; e < NEXP; ++e) { lg[e] = acc[e] + bg[e]; mx = fmaxf(mx, lg[e]); }
        float p[NEXP];
#pragma unroll
        for (int e = 0; e < NEXP; ++e) p[e] = expf(lg[e] - mx);
        bool taken[NEXP];
#pragma unroll
        for (int e = 0; e < NEXP; ++e) taken[e] = false;
        float stop = 0.f;
        int   sel[KTOP];
#pragma unroll
        for (int k = 0; k < KTOP; ++k) {
            int   bi = -1;
            float bv = -1.f;
            for (int e = 0; e < NEXP; ++e)
                if (!taken[e] && p[e] > bv) { bv = p[e]; bi = e; }
            taken[bi] = true; sel[k] = bi; stop += bv;
        }
        float inv = 1.f / stop;
        float c[NEXP];
#pragma unroll
        for (int e = 0; e < NEXP; ++e) c[e] = 0.f;
#pragma unroll
        for (int k = 0; k < KTOP; ++k) c[sel[k]] = p[sel[k]] * inv;
        float* lo = logits_out + (size_t)t * NEXP;
        float* co = comb + (size_t)t * NEXP;
#pragma unroll
        for (int e = 0; e < NEXP; ++e) { lo[e] = lg[e]; co[e] = c[e]; }
    }
}

// ---------------------------------------------------------------------------
// WMMA fragment helper (CDNA5 wave32, 16-bit, 16x16x32):
//   A (16x32 MxK): lane L -> M=L&15; lanes<16: K=0..7,16..23; lanes>=16:
//                  K=8..15,24..31.
//   B (32x16 KxN): lane L -> N=L&15; lanes<16: K=0..7,16..23; lanes>=16: K=8..15,24..31
//                  (same K order as A, via transposed LDS tile -> 2x ds_load_b128).
//   C/D: VGPR r, lanes<16 -> M=r, lanes>=16 -> M=r+8; N=L&15.
// ---------------------------------------------------------------------------
__device__ inline v16bf cat8(v8bf lo, v8bf hi)
{
    return __builtin_shufflevector(lo, hi, 0, 1, 2, 3, 4, 5, 6, 7,
                                   8, 9, 10, 11, 12, 13, 14, 15);
}

union accu { v8f v; float f[8]; };

// ---------------------------------------------------------------------------
// Unified expert-FFN GEMM. MODE 0: hbuf = gelu_exact(A @ W + bias) (bf16 out)
//                          MODE 1: out += comb[:,e] * (A @ W + bias) (fp32)
// Block tile 128x128, BK=32, 8 waves (4 M x 2 N), each wave 32x64 = 8 WMMAs.
// Double-buffered LDS: one s_barrier per K-step, globals overlap WMMA.
// ---------------------------------------------------------------------------
template <int MODE>
__global__ __launch_bounds__(256) void moe_ffn(
    const bf16_t* __restrict__ Abf,  // [T, 2048] bf16 (xbf or hbuf)
    const float*  __restrict__ W,    // [2048, 2048] fp32 expert slice
    const float*  __restrict__ bias, // [2048]
    const float*  __restrict__ comb, // [T, E]
    int e,
    bf16_t* __restrict__ hbuf,       // MODE 0 target
    float*  __restrict__ out)        // MODE 1 target
{
    constexpr int KD = 2048;  // reduction dim (D for ffn1, H for ffn2)
    constexpr int ND = 2048;  // output dim   (H for ffn1, D for ffn2)

    __shared__ bf16_t As[2][128][32];   // [buf][row][k]
    __shared__ bf16_t Bs[2][128][32];   // [buf][n][k]  (transposed)
    __shared__ int    active;

    const int tid = threadIdx.x;
    const int rowBase = blockIdx.x * 128;
    const int colBase = blockIdx.y * 128;

    if (tid == 0) active = 0;
    __syncthreads();
    if (tid < 128 && comb[(size_t)(rowBase + tid) * NEXP + e] != 0.f)
        atomicOr(&active, 1);
    __syncthreads();
    if (!active) return;   // uniform across the block

    const int wid = tid >> 5, lane = tid & 31;
    const int wm = wid & 3;   // 0..3 -> 32-row slab
    const int wn = wid >> 2;  // 0..1 -> 64-col slab

    v8f acc[2][4] = {};

    // staging maps: A = 256 thr x (1 row x 16 k);  B = 256 thr x (4 k x 4 n)
    const int sa_row = tid >> 1;          // 0..127
    const int sa_k   = (tid & 1) * 16;    // 0,16
    const int sb_kq  = (tid >> 5) * 4;    // 0,4,...,28
    const int sb_n   = (tid & 31) * 4;    // 0,4,...,124

    const bf16_t* agp = Abf + (size_t)(rowBase + sa_row) * KD + sa_k;
    const float*  bgp = W + (size_t)sb_kq * ND + colBase + sb_n;

    // fragment coordinates
    const int fr  = lane & 15;
    const int fka = (lane < 16) ? 0 : 8;
    const int fkb = (lane < 16) ? 0 : 8;

    // ---- prologue: stage K-step 0 into buffer 0 ----
    {
        uint4 a0 = *(const uint4*)agp;
        uint4 a1 = *(const uint4*)(agp + 8);
        float4 r0 = *(const float4*)bgp;
        float4 r1 = *(const float4*)(bgp + ND);
        float4 r2 = *(const float4*)(bgp + 2 * ND);
        float4 r3 = *(const float4*)(bgp + 3 * ND);
        agp += 32; bgp += (size_t)32 * ND;
        *(uint4*)&As[0][sa_row][sa_k]     = a0;
        *(uint4*)&As[0][sa_row][sa_k + 8] = a1;
        v4bf c0 = { (bf16_t)r0.x, (bf16_t)r1.x, (bf16_t)r2.x, (bf16_t)r3.x };
        v4bf c1 = { (bf16_t)r0.y, (bf16_t)r1.y, (bf16_t)r2.y, (bf16_t)r3.y };
        v4bf c2 = { (bf16_t)r0.z, (bf16_t)r1.z, (bf16_t)r2.z, (bf16_t)r3.z };
        v4bf c3 = { (bf16_t)r0.w, (bf16_t)r1.w, (bf16_t)r2.w, (bf16_t)r3.w };
        *(v4bf*)&Bs[0][sb_n + 0][sb_kq] = c0;
        *(v4bf*)&Bs[0][sb_n + 1][sb_kq] = c1;
        *(v4bf*)&Bs[0][sb_n + 2][sb_kq] = c2;
        *(v4bf*)&Bs[0][sb_n + 3][sb_kq] = c3;
    }
    __syncthreads();

    constexpr int NK = KD / 32;  // 64
    for (int kt = 0; kt < NK; ++kt) {
        const int  cur  = kt & 1;
        const int  nxt  = cur ^ 1;
        const bool more = (kt + 1) < NK;

        // issue next K-step's global loads (overlap with WMMA below)
        uint4 a0, a1; float4 r0, r1, r2, r3;
        if (more) {
            a0 = *(const uint4*)agp;
            a1 = *(const uint4*)(agp + 8);
            r0 = *(const float4*)bgp;
            r1 = *(const float4*)(bgp + ND);
            r2 = *(const float4*)(bgp + 2 * ND);
            r3 = *(const float4*)(bgp + 3 * ND);
            __builtin_prefetch(bgp + (size_t)32 * ND, 0, 3);
            agp += 32; bgp += (size_t)32 * ND;
        }

        // fragments from current buffer
        v16bf af[2];
#pragma unroll
        for (int m = 0; m < 2; ++m) {
            const int row = wm * 32 + m * 16 + fr;
            af[m] = cat8(*(const v8bf*)&As[cur][row][fka],
                         *(const v8bf*)&As[cur][row][fka + 16]);
        }
        v16bf bfr[4];
#pragma unroll
        for (int j = 0; j < 4; ++j) {
            const int col = wn * 64 + j * 16 + fr;
            bfr[j] = cat8(*(const v8bf*)&Bs[cur][col][fkb],
                          *(const v8bf*)&Bs[cur][col][fkb + 16]);
        }

        // 8 WMMAs: 2 (M) x 4 (N) outer product
#pragma unroll
        for (int m = 0; m < 2; ++m)
#pragma unroll
            for (int j = 0; j < 4; ++j)
                acc[m][j] = __builtin_amdgcn_wmma_f32_16x16x32_bf16(
                    false, af[m], false, bfr[j], (short)0, acc[m][j],
                    false, false);

        // stage next K-step into the other buffer
        if (more) {
            *(uint4*)&As[nxt][sa_row][sa_k]     = a0;
            *(uint4*)&As[nxt][sa_row][sa_k + 8] = a1;
            v4bf c0 = { (bf16_t)r0.x, (bf16_t)r1.x, (bf16_t)r2.x, (bf16_t)r3.x };
            v4bf c1 = { (bf16_t)r0.y, (bf16_t)r1.y, (bf16_t)r2.y, (bf16_t)r3.y };
            v4bf c2 = { (bf16_t)r0.z, (bf16_t)r1.z, (bf16_t)r2.z, (bf16_t)r3.z };
            v4bf c3 = { (bf16_t)r0.w, (bf16_t)r1.w, (bf16_t)r2.w, (bf16_t)r3.w };
            *(v4bf*)&Bs[nxt][sb_n + 0][sb_kq] = c0;
            *(v4bf*)&Bs[nxt][sb_n + 1][sb_kq] = c1;
            *(v4bf*)&Bs[nxt][sb_n + 2][sb_kq] = c2;
            *(v4bf*)&Bs[nxt][sb_n + 3][sb_kq] = c3;
        }
        __syncthreads();
    }

    // ---- epilogue ----
    const int moff = (lane < 16) ? 0 : 8;
#pragma unroll
    for (int m = 0; m < 2; ++m) {
#pragma unroll
        for (int j = 0; j < 4; ++j) {
            accu u; u.v = acc[m][j];
            const int n  = colBase + wn * 64 + j * 16 + fr;
            const float bb = bias[n];
#pragma unroll
            for (int r = 0; r < 8; ++r) {
                const int mm = rowBase + wm * 32 + m * 16 + moff + r;
                float v = u.f[r] + bb;
                if (MODE == 0) {
                    v = 0.5f * v * (1.f + erff(v * 0.70710678118654752f));
                    hbuf[(size_t)mm * HDIM + n] = (bf16_t)v;
                } else {
                    float w = comb[(size_t)mm * NEXP + e];
                    if (w != 0.f)
                        out[(size_t)mm * DDIM + n] += w * v;
                }
            }
        }
    }
}

// ---------------------------------------------------------------------------
// launch
// ---------------------------------------------------------------------------
extern "C" void kernel_launch(void* const* d_in, const int* in_sizes, int n_in,
                              void* d_out, int out_size, void* d_ws, size_t ws_size,
                              hipStream_t stream)
{
    const float* x  = (const float*)d_in[0];   // [T, D]
    const float* Wg = (const float*)d_in[1];   // [D, E]
    const float* bg = (const float*)d_in[2];   // [E]
    const float* W1 = (const float*)d_in[3];   // [E, D, H]
    const float* b1 = (const float*)d_in[4];   // [E, H]
    const float* W2 = (const float*)d_in[5];   // [E, H, D]
    const float* b2 = (const float*)d_in[6];   // [E, D]

    float* out_final  = (float*)d_out;                      // [T, D]
    float* out_logits = out_final + (size_t)T_TOK * DDIM;   // [T, E]

    // workspace layout (16B-aligned offsets)
    char* ws = (char*)d_ws;
    float*  comb = (float*)ws;                                        // T*E f32
    bf16_t* xbf  = (bf16_t*)(ws + 131072);                            // T*D bf16
    bf16_t* hbuf = (bf16_t*)(ws + 131072 + (size_t)T_TOK * DDIM * 2); // T*H bf16

    // 1) x -> bf16 (read once, reused by all experts)
    cvt_f32_to_bf16<<<2048, 256, 0, stream>>>(x, xbf, (T_TOK * DDIM) / 4);

    // 2) routing (raw logits to d_out tail + combine weights to ws)
    moe_route<<<T_TOK / 8, 256, 0, stream>>>(x, Wg, bg, out_logits, comb);

    // 3) zero accumulation target
    zero_f32<<<2048, 256, 0, stream>>>(out_final, (T_TOK * DDIM) / 4);

    // 4) per-expert FFN1 -> FFN2+combine (serialized on stream => plain +=)
    dim3 grid1(T_TOK / 128, HDIM / 128);   // 32 x 16
    dim3 grid2(T_TOK / 128, DDIM / 128);   // 32 x 16
    for (int e = 0; e < NEXP; ++e) {
        const float* W1e = W1 + (size_t)e * DDIM * HDIM;
        const float* b1e = b1 + (size_t)e * HDIM;
        const float* W2e = W2 + (size_t)e * HDIM * DDIM;
        const float* b2e = b2 + (size_t)e * DDIM;
        moe_ffn<0><<<grid1, 256, 0, stream>>>(xbf, W1e, b1e, comb, e, hbuf, out_final);
        moe_ffn<1><<<grid2, 256, 0, stream>>>(hbuf, W2e, b2e, comb, e, hbuf, out_final);
    }
}
